// HydraLoRAModule_62749472194694
// MI455X (gfx1250) — matrix operand order repaired and checked by reading, not compiled, not run on hardware
//
#include <hip/hip_runtime.h>

#define B_    4
#define S_    4096
#define DIN_  2048
#define DOUT_ 2048
#define R_    16
#define E_    12
#define SCALE_MULT 1.0f   // MULTIPLIER * SCALE

typedef __bf16 bf16x16 __attribute__((ext_vector_type(16)));
typedef float  v8f     __attribute__((ext_vector_type(8)));
typedef unsigned int u32x4 __attribute__((ext_vector_type(4)));
typedef int          i32x4 __attribute__((ext_vector_type(4)));
typedef int          i32x8 __attribute__((ext_vector_type(8)));

union FragU { uint4 q[2]; bf16x16 v; };

// LDS tile geometry: 32-half rows padded to 40 halfs (80 B) -> conflict-free
// b128 fragment reads (gcd(20 banks,64)=4, 16-lane cycle covers 16 distinct banks).
#define LDS_STRIDE 40
#define A_ROWS     256
#define B_ROWS     128
#define A_HALFS    (A_ROWS * LDS_STRIDE)   // 10240 halfs = 20480 B
#define B_HALFS    (B_ROWS * LDS_STRIDE)   //  5120 halfs = 10240 B
#define BUF_HALFS  (A_HALFS + B_HALFS)     // 15360 halfs = 30720 B per buffer

#if __has_builtin(__builtin_amdgcn_tensor_load_to_lds) && __has_builtin(__builtin_amdgcn_s_wait_tensorcnt)
#define USE_TDM 1
#else
#define USE_TDM 0
#endif

__device__ __forceinline__ unsigned short f2bf(float f) {
    union { float f; unsigned int u; } c; c.f = f;
    unsigned int u = c.u;
    u += 0x7FFFu + ((u >> 16) & 1u);      // round-to-nearest-even
    return (unsigned short)(u >> 16);
}

__device__ __forceinline__ unsigned lds_addr_of(const void* p) {
    return (unsigned)(unsigned long long)(__attribute__((address_space(3))) const void*)p;
}

#if USE_TDM
// Build a 2D-tile D# (type=2, count=1, data_size=2B) with LDS padding:
// pad_interval code 3 -> every 16 DWORDs (one 32-half row), pad_amount code 3 -> 4 DWORDs.
// Net LDS row stride = 32 + 8 halfs = LDS_STRIDE.
__device__ __forceinline__ void tdm_load_tile(unsigned lds_byte, const unsigned short* g,
                                              unsigned tdim0, unsigned tdim1,
                                              unsigned tile0, unsigned tile1,
                                              unsigned stride0) {
    unsigned long long ga = (unsigned long long)g;
    u32x4 g0;
    g0.x = 1u;                                                  // count=1 (valid user D#)
    g0.y = lds_byte;                                            // lds_addr
    g0.z = (unsigned)ga;                                        // global_addr[31:0]
    g0.w = (unsigned)((ga >> 32) & 0x01FFFFFFu) | (2u << 30);   // addr[56:32] | type=2
    i32x8 g1;
    g1[0] = (int)((1u << 16) | (1u << 20) | (3u << 22) | (3u << 25)); // data_size=2B,pad_en,intv=16dw,amt=4dw
    g1[1] = (int)((tdim0 & 0xFFFFu) << 16);                     // tensor_dim0[15:0]
    g1[2] = (int)((tdim0 >> 16) | ((tdim1 & 0xFFFFu) << 16));   // dim0[31:16] | dim1[15:0]
    g1[3] = (int)((tdim1 >> 16) | ((tile0 & 0xFFFFu) << 16));   // dim1[31:16] | tile_dim0
    g1[4] = (int)(tile1 & 0xFFFFu);                             // tile_dim1 (tile_dim2=0)
    g1[5] = (int)stride0;                                       // tensor_dim0_stride[31:0]
    g1[6] = 0;
    g1[7] = 0;
    i32x4 z4 = {0, 0, 0, 0};
#if __has_include(<hip/amd_detail/amd_gfx1250_TDM.h>)
    i32x8 z8 = {0, 0, 0, 0, 0, 0, 0, 0};
    __builtin_amdgcn_tensor_load_to_lds(g0, g1, z4, z4, z8, 0); // 6-arg toolchain
#else
    __builtin_amdgcn_tensor_load_to_lds(g0, g1, z4, z4, 0);     // 5-arg ROCm 7.2
#endif
}
#endif

// ---------------------------------------------------------------- convert fp32 -> bf16
__global__ void cvt_f32_bf16_kernel(const float* __restrict__ src,
                                    unsigned short* __restrict__ dst, int n4) {
    int i = blockIdx.x * blockDim.x + threadIdx.x;
    if (i >= n4) return;
    float4 f = ((const float4*)src)[i];
    ushort4 o;
    o.x = f2bf(f.x); o.y = f2bf(f.y); o.z = f2bf(f.z); o.w = f2bf(f.w);
    ((ushort4*)dst)[i] = o;
}

// ---------------------------------------------------------------- combined = gate @ W_up (pre-scaled), bf16 [B][DOUT][R]
__global__ void combined_kernel(const float* __restrict__ gate,
                                const float* __restrict__ W_up,
                                unsigned short* __restrict__ comb) {
    int idx = blockIdx.x * blockDim.x + threadIdx.x;       // b*DOUT + o
    if (idx >= B_ * DOUT_) return;
    int b = idx / DOUT_;
    int o = idx - b * DOUT_;
    float acc[R_];
#pragma unroll
    for (int r = 0; r < R_; ++r) acc[r] = 0.0f;
    for (int e = 0; e < E_; ++e) {
        float g = gate[b * E_ + e];
        const float* w = W_up + ((size_t)e * DOUT_ + o) * R_;
#pragma unroll
        for (int r = 0; r < R_; ++r) acc[r] += g * w[r];
    }
    unsigned short* dst = comb + (size_t)idx * R_;
#pragma unroll
    for (int r = 0; r < R_; ++r) dst[r] = f2bf(acc[r] * SCALE_MULT);
}

// ---------------------------------------------------------------- lx = x @ W_down^T  (WMMA, N=16)
__global__ __launch_bounds__(128) void lx_kernel(const unsigned short* __restrict__ xbf,
                                                 const unsigned short* __restrict__ wdn,
                                                 unsigned short* __restrict__ lx) {
    const int lane = threadIdx.x & 31, sub = lane & 15, hi = lane >> 4;
    const int wave = blockIdx.x * 4 + (threadIdx.x >> 5);
    const int m0   = wave * 64;
    v8f acc[4] = {};
    for (int k0 = 0; k0 < DIN_; k0 += 32) {
        FragU bb;
        const uint4* bp = (const uint4*)(wdn + (size_t)sub * DIN_ + k0 + hi * 8);
        bb.q[0] = bp[0]; bb.q[1] = bp[2];
#pragma unroll
        for (int mi = 0; mi < 4; ++mi) {
            FragU a;
            const unsigned short* ap = xbf + (size_t)(m0 + mi * 16 + sub) * DIN_ + k0 + hi * 8;
            a.q[0] = *(const uint4*)ap;
            a.q[1] = *(const uint4*)(ap + 16);
            acc[mi] = __builtin_amdgcn_wmma_f32_16x16x32_bf16(
                false, a.v, false, bb.v, (short)0, acc[mi], false, false);
        }
    }
#pragma unroll
    for (int mi = 0; mi < 4; ++mi)
#pragma unroll
        for (int v = 0; v < 8; ++v)
            lx[(size_t)(m0 + mi * 16 + v + 8 * hi) * R_ + sub] = f2bf(acc[mi][v]);
}

// ---------------------------------------------------------------- main: out = x@W_org^T + lx@combined^T
// 256 threads = 8 waves (4M x 2N), block tile 256x128, LDS double-buffered via TDM.
__global__ __launch_bounds__(256) void hydra_gemm_kernel(const unsigned short* __restrict__ xbf,
                                                         const unsigned short* __restrict__ wbf,
                                                         const unsigned short* __restrict__ lx,
                                                         const unsigned short* __restrict__ comb,
                                                         float* __restrict__ out) {
    __shared__ __align__(16) unsigned short smem[2 * BUF_HALFS];   // 61440 B

    const int lane = threadIdx.x & 31, sub = lane & 15, hi = lane >> 4;
    const int wave = threadIdx.x >> 5;
    const int wm = wave & 3, wn = wave >> 2;
    const int m_block = blockIdx.y * 256;
    const int n_block = blockIdx.x * 128;
    const int m_wave = wm * 64;      // block-local
    const int n_wave = wn * 64;
    const int NK = DIN_ / 32;

    v8f acc[4][4] = {};

#if USE_TDM
    const unsigned lds_base = lds_addr_of(smem);
    if (wave == 0) {
        tdm_load_tile(lds_base, xbf + (size_t)m_block * DIN_,
                      DIN_, B_ * S_, 32, A_ROWS, DIN_);
        tdm_load_tile(lds_base + A_HALFS * 2, wbf + (size_t)n_block * DIN_,
                      DIN_, DOUT_, 32, B_ROWS, DIN_);
        __builtin_amdgcn_s_wait_tensorcnt(0);
    }
    __syncthreads();
#endif

    for (int it = 0; it < NK; ++it) {
        const unsigned cur = (unsigned)(it & 1) * BUF_HALFS;

#if USE_TDM
        if ((it + 1 < NK) && wave == 0) {      // prefetch next K tile into other buffer
            const unsigned nxt = (unsigned)((it + 1) & 1) * BUF_HALFS;
            const int k0n = (it + 1) * 32;
            tdm_load_tile(lds_base + nxt * 2, xbf + (size_t)m_block * DIN_ + k0n,
                          DIN_, B_ * S_, 32, A_ROWS, DIN_);
            tdm_load_tile(lds_base + (nxt + A_HALFS) * 2, wbf + (size_t)n_block * DIN_ + k0n,
                          DIN_, DOUT_, 32, B_ROWS, DIN_);
        }
#else
        { // synchronous fallback: cooperative global->LDS copy of buffer `cur`
            const int k0 = it * 32;
            __syncthreads();
#pragma unroll
            for (int t = 0; t < 4; ++t) {      // A: 1024 x 16B chunks
                int c = threadIdx.x + t * 256;
                int row = c >> 2, seg = (c & 3) * 8;
                *(uint4*)(smem + cur + row * LDS_STRIDE + seg) =
                    *(const uint4*)(xbf + (size_t)(m_block + row) * DIN_ + k0 + seg);
            }
#pragma unroll
            for (int t = 0; t < 2; ++t) {      // B: 512 x 16B chunks
                int c = threadIdx.x + t * 256;
                int row = c >> 2, seg = (c & 3) * 8;
                *(uint4*)(smem + cur + A_HALFS + row * LDS_STRIDE + seg) =
                    *(const uint4*)(wbf + (size_t)(n_block + row) * DIN_ + k0 + seg);
            }
            __syncthreads();
        }
#endif

        bf16x16 af[4], bfv[4];
#pragma unroll
        for (int mi = 0; mi < 4; ++mi) {       // A frags (ds_load_b128 x2)
            FragU a;
            const unsigned off = cur + (unsigned)(m_wave + mi * 16 + sub) * LDS_STRIDE + hi * 8;
            a.q[0] = *(const uint4*)(smem + off);
            a.q[1] = *(const uint4*)(smem + off + 16);
            af[mi] = a.v;
        }
#pragma unroll
        for (int ni = 0; ni < 4; ++ni) {       // B frags (ds_load_b128 x2)
            FragU bb;
            const unsigned off = cur + A_HALFS + (unsigned)(n_wave + ni * 16 + sub) * LDS_STRIDE + hi * 8;
            bb.q[0] = *(const uint4*)(smem + off);
            bb.q[1] = *(const uint4*)(smem + off + 16);
            bfv[ni] = bb.v;
        }
#pragma unroll
        for (int mi = 0; mi < 4; ++mi)
#pragma unroll
            for (int ni = 0; ni < 4; ++ni)
                acc[mi][ni] = __builtin_amdgcn_wmma_f32_16x16x32_bf16(
                    false, af[mi], false, bfv[ni], (short)0, acc[mi][ni], false, false);

#if USE_TDM
        if (wave == 0) __builtin_amdgcn_s_wait_tensorcnt(0);
        __syncthreads();                        // next-buffer data ready; cur free to overwrite
#endif
    }

    // ---- rank-16 LoRA correction: one zero-padded WMMA per 16x16 tile
    const int batch = m_block >> 12;            // / S_; 256-row block never straddles batches
    const uint4 z = make_uint4(0, 0, 0, 0);
    bf16x16 lxf[4], cbf[4];
#pragma unroll
    for (int mi = 0; mi < 4; ++mi) {
        FragU a;
        a.q[0] = *(const uint4*)(lx + (size_t)(m_block + m_wave + mi * 16 + sub) * R_ + hi * 8);
        a.q[1] = z;                             // K=16..31 zero-padded
        lxf[mi] = a.v;
    }
#pragma unroll
    for (int ni = 0; ni < 4; ++ni) {
        FragU bb;
        const uint4* bp = (const uint4*)(comb +
            ((size_t)batch * DOUT_ + n_block + n_wave + ni * 16 + sub) * R_);
        bb.q[0] = bp[hi];                       // K=8*hi..8*hi+7
        bb.q[1] = z;                            // K=16..31 zero-padded
        cbf[ni] = bb.v;
    }
#pragma unroll
    for (int mi = 0; mi < 4; ++mi)
#pragma unroll
        for (int ni = 0; ni < 4; ++ni)
            acc[mi][ni] = __builtin_amdgcn_wmma_f32_16x16x32_bf16(
                false, lxf[mi], false, cbf[ni], (short)0, acc[mi][ni], false, false);

    // ---- store fp32 per C-matrix layout (VGPR v -> M=v / v+8, lane&15 -> N)
#pragma unroll
    for (int mi = 0; mi < 4; ++mi)
#pragma unroll
        for (int ni = 0; ni < 4; ++ni)
#pragma unroll
            for (int v = 0; v < 8; ++v)
                out[(size_t)(m_block + m_wave + mi * 16 + v + 8 * hi) * DOUT_
                    + n_block + n_wave + ni * 16 + sub] = acc[mi][ni][v];
}

// ---------------------------------------------------------------- launch
extern "C" void kernel_launch(void* const* d_in, const int* in_sizes, int n_in,
                              void* d_out, int out_size, void* d_ws, size_t ws_size,
                              hipStream_t stream) {
    const float* x      = (const float*)d_in[0];
    const float* gate   = (const float*)d_in[1];
    const float* W_org  = (const float*)d_in[2];
    const float* W_down = (const float*)d_in[3];
    const float* W_up   = (const float*)d_in[4];
    float* out = (float*)d_out;

    char* ws = (char*)d_ws;
    unsigned short* xbf  = (unsigned short*)(ws + 0);          // 67,108,864 B
    unsigned short* wbf  = (unsigned short*)(ws + 67108864);   //  8,388,608 B
    unsigned short* wdn  = (unsigned short*)(ws + 75497472);   //     65,536 B
    unsigned short* lx   = (unsigned short*)(ws + 75563008);   //    524,288 B
    unsigned short* comb = (unsigned short*)(ws + 76087296);   //    262,144 B

    {
        int n4 = (B_ * S_ * DIN_) / 4;
        cvt_f32_bf16_kernel<<<(n4 + 255) / 256, 256, 0, stream>>>(x, xbf, n4);
    }
    {
        int n4 = (DOUT_ * DIN_) / 4;
        cvt_f32_bf16_kernel<<<(n4 + 255) / 256, 256, 0, stream>>>(W_org, wbf, n4);
    }
    {
        int n4 = (R_ * DIN_) / 4;
        cvt_f32_bf16_kernel<<<(n4 + 255) / 256, 256, 0, stream>>>(W_down, wdn, n4);
    }
    combined_kernel<<<(B_ * DOUT_ + 255) / 256, 256, 0, stream>>>(gate, W_up, comb);
    lx_kernel<<<64, 128, 0, stream>>>(xbf, wdn, lx);
    hydra_gemm_kernel<<<dim3(DOUT_ / 128, (B_ * S_) / 256), 256, 0, stream>>>(
        xbf, wbf, lx, comb, out);
}
